// MiniBatchEdgePropPlus_35665408425987
// MI455X (gfx1250) — matrix operand order repaired and checked by reading, not compiled, not run on hardware
//
#include <hip/hip_runtime.h>
#include <math.h>

#define HDIM 256

typedef __attribute__((ext_vector_type(16))) __bf16 v16bf;
typedef __attribute__((ext_vector_type(8)))  float  v8f;

// ---------------------------------------------------------------------------
// WMMA fragment builders (CDNA5 16x16x32 bf16 layouts, wave32)
// ---------------------------------------------------------------------------

// A-matrix 16x32 bf16 from an LDS bf16 tile (row-major, leading dim ld).
// lane&15 = row, lane>>4 = half. half0 covers K {0..15}, half1 {16..31}.
// Two contiguous 8-element (16B) chunks -> 2x ds_load_b128, no conversions.
__device__ __forceinline__ v16bf a_frag_lds_bf(const __bf16* A, int ld, int k0, int lane) {
  const int row  = lane & 15;
  const int half = lane >> 4;
  const __bf16* p = A + row * ld + k0 + 16 * half;
  v16bf a;
#pragma unroll
  for (int q = 0; q < 8; ++q) a[q] = p[q];
#pragma unroll
  for (int q = 8; q < 16; ++q) a[q] = p[q];
  return a;
}

// B-matrix 32x16 bf16 from weight matrix W[N,K] (row-major bf16), computing W^T.
// lane&15 = output column n, halves split K at 16. 16 contiguous bf16 per lane
// -> 2x global_load_b128.
__device__ __forceinline__ v16bf b_frag_w(const __bf16* W, int ldk, int n0, int k0, int lane) {
  const int n    = n0 + (lane & 15);
  const int half = lane >> 4;
  const __bf16* p = W + (size_t)n * ldk + k0 + 16 * half;
  v16bf b;
#pragma unroll
  for (int q = 0; q < 16; ++q) b[q] = p[q];
  return b;
}

__device__ __forceinline__ float sigmoidf_(float x) {
  return 1.0f / (1.0f + __expf(-x));
}

// ---------------------------------------------------------------------------
// fp32 -> bf16 weight conversion (weights then live in the 192MB L2)
// ---------------------------------------------------------------------------
__global__ void cvt_bf16_kernel(const float* __restrict__ src, __bf16* __restrict__ dst, int n) {
  int i = blockIdx.x * 256 + threadIdx.x;
  if (i < n) dst[i] = (__bf16)src[i];
}

__global__ void zero_f32_kernel(float* __restrict__ p, int n) {
  int i = blockIdx.x * 256 + threadIdx.x;
  if (i < n) p[i] = 0.0f;
}

// ---------------------------------------------------------------------------
// GRU over T timesteps, mean of outputs.  16 edges per block, 512 threads =
// 16 wave32s; wave w owns the single 16-channel sub-tile 16w..16w+15 of all
// three gates (r/z/n) -> gate math is wave-local and only 6 accumulators are
// live (VGPRs stay well under 256).  The h recurrence value lives in per-lane
// registers (fp32); LDS holds only the bf16 copy consumed as the WMMA A side.
// ---------------------------------------------------------------------------
__global__ __launch_bounds__(512) void gru_mean_kernel(
    const float* __restrict__ x, int E, int T, int Kin,
    const __bf16* __restrict__ Wih, const __bf16* __restrict__ Whh,
    const float* __restrict__ bih, const float* __restrict__ bhh,
    float* __restrict__ out) {
  __shared__ __bf16 hbuf[16 * HDIM];  // 8 KB
  __shared__ __bf16 xbuf[16 * HDIM];  // 8 KB

  const int tid  = threadIdx.x;
  const int lane = tid & 31;
  const int w    = tid >> 5;   // 0..15
  const int e0   = blockIdx.x * 16;
  const int c0   = w * 16;     // this wave's 16 channels
  const int half = lane >> 4;
  const int nlo  = lane & 15;

  for (int i = tid; i < 16 * HDIM; i += 512) hbuf[i] = (__bf16)0.0f;

  v8f hprev = {0.f, 0.f, 0.f, 0.f, 0.f, 0.f, 0.f, 0.f};
  v8f hsum  = {0.f, 0.f, 0.f, 0.f, 0.f, 0.f, 0.f, 0.f};

  const int ksteps_i = Kin >> 5;

  for (int t = 0; t < T; ++t) {
    // stage x[:, t, :] tile into LDS as bf16 (cvt once, reused by all waves)
    for (int i = tid; i < 16 * Kin; i += 512) {
      int r = i / Kin, c = i - r * Kin;
      int e = e0 + r;
      xbuf[r * HDIM + c] = (e < E) ? (__bf16)x[((size_t)e * T + t) * Kin + c] : (__bf16)0.0f;
    }
    // prefetch next timestep's rows while this step computes
    if (t + 1 < T && tid < 16 && (e0 + tid) < E)
      __builtin_prefetch(x + ((size_t)(e0 + tid) * T + t + 1) * Kin, 0, 0);
    __syncthreads();  // orders previous hbuf writes + xbuf staging before GEMMs

    // acc[0..2] = gi gates (r,z,n), acc[3..5] = gh gates (r,z,n)
    v8f acc[6];
#pragma unroll
    for (int g = 0; g < 3; ++g) {
      float bi = bih[g * HDIM + c0 + nlo];
      float bh = bhh[g * HDIM + c0 + nlo];
      acc[g]     = {bi, bi, bi, bi, bi, bi, bi, bi};
      acc[3 + g] = {bh, bh, bh, bh, bh, bh, bh, bh};
    }
    // gi = x_t @ Wih^T : one A-fragment + 3 prefetched B-fragments per k-step
    for (int kk = 0; kk < ksteps_i; ++kk) {
      v16bf a  = a_frag_lds_bf(xbuf, HDIM, kk * 32, lane);
      v16bf b0 = b_frag_w(Wih, Kin, 0 * HDIM + c0, kk * 32, lane);
      v16bf b1 = b_frag_w(Wih, Kin, 1 * HDIM + c0, kk * 32, lane);
      v16bf b2 = b_frag_w(Wih, Kin, 2 * HDIM + c0, kk * 32, lane);
      acc[0] = __builtin_amdgcn_wmma_f32_16x16x32_bf16(false, a, false, b0, (short)0, acc[0],
                                                       false, false);
      acc[1] = __builtin_amdgcn_wmma_f32_16x16x32_bf16(false, a, false, b1, (short)0, acc[1],
                                                       false, false);
      acc[2] = __builtin_amdgcn_wmma_f32_16x16x32_bf16(false, a, false, b2, (short)0, acc[2],
                                                       false, false);
    }
    // gh = h @ Whh^T
#pragma unroll
    for (int kk = 0; kk < HDIM / 32; ++kk) {
      v16bf a  = a_frag_lds_bf(hbuf, HDIM, kk * 32, lane);
      v16bf b0 = b_frag_w(Whh, HDIM, 0 * HDIM + c0, kk * 32, lane);
      v16bf b1 = b_frag_w(Whh, HDIM, 1 * HDIM + c0, kk * 32, lane);
      v16bf b2 = b_frag_w(Whh, HDIM, 2 * HDIM + c0, kk * 32, lane);
      acc[3] = __builtin_amdgcn_wmma_f32_16x16x32_bf16(false, a, false, b0, (short)0, acc[3],
                                                       false, false);
      acc[4] = __builtin_amdgcn_wmma_f32_16x16x32_bf16(false, a, false, b1, (short)0, acc[4],
                                                       false, false);
      acc[5] = __builtin_amdgcn_wmma_f32_16x16x32_bf16(false, a, false, b2, (short)0, acc[5],
                                                       false, false);
    }
    // wave-local gate math; h_old comes from this lane's own registers
    v8f hnew;
#pragma unroll
    for (int i = 0; i < 8; ++i) {
      float r  = sigmoidf_(acc[0][i] + acc[3][i]);
      float z  = sigmoidf_(acc[1][i] + acc[4][i]);
      float nn = tanhf(acc[2][i] + r * acc[5][i]);
      float hn = (1.0f - z) * nn + z * hprev[i];
      hnew[i] = hn;
      hsum[i] += hn;
    }
    __syncthreads();  // every wave finished reading hbuf
    const int ch = c0 + nlo;
#pragma unroll
    for (int i = 0; i < 8; ++i) {
      hbuf[(i + 8 * half) * HDIM + ch] = (__bf16)hnew[i];
      hprev[i] = hnew[i];
    }
  }

  const float invT = 1.0f / (float)T;
  const int ch = c0 + nlo;
#pragma unroll
  for (int i = 0; i < 8; ++i) {
    const int e = e0 + i + 8 * half;
    if (e < E) out[(size_t)e * HDIM + ch] = hsum[i] * invT;
  }
}

// ---------------------------------------------------------------------------
// Generic WMMA GEMM on a gathered/concatenated A:
//   A[m, 0:cols1)           = A1 ? A1[idx1?idx1[m]:m, :] : zeros
//   A[m, cols1:cols1+cols2) = A2[idx2?idx2[m]:m, :]
//   raw = A @ W^T + bias          (W is [N,K] bf16, K = cols1+cols2)
// mode 0: Out[m,n] = raw                                  (any N, no LN)
// mode 1: Out[m,c] = relu(LN(raw)*gamma+beta)             (N == 256)
// mode 2: v = relu(LN(raw)*gamma+beta);
//         msg = relu(v - hist[histIdx[m], c]);
//         atomicAdd(Out[dstIdx[m], c], msg)               (N == 256)
// 16 rows/block, 8 waves; wave w owns 16-col tiles {w, w+8, ...}, NT per wave
// (compile-time so accumulators stay in registers).  Tile-validity tests are
// wave-uniform -> EXEC stays all-ones for WMMA.
// ---------------------------------------------------------------------------
template <int NT>
__global__ __launch_bounds__(256) void gemm_ln_kernel(
    const float* __restrict__ A1, const int* __restrict__ idx1, int cols1,
    const float* __restrict__ A2, const int* __restrict__ idx2, int cols2,
    const __bf16* __restrict__ W, const float* __restrict__ bias,
    int M, int N, int mode,
    const float* __restrict__ gamma, const float* __restrict__ beta,
    const float* __restrict__ hist, const int* __restrict__ histIdx,
    const int* __restrict__ dstIdx,
    float* __restrict__ Out) {
  __shared__ __bf16 Ash[16 * 512];   // 16 KB
  __shared__ float outsh[16 * HDIM]; // 16 KB
  __shared__ float mu_s[16], rs_s[16];

  const int tid  = threadIdx.x;
  const int lane = tid & 31;
  const int w    = tid >> 5;
  const int m0   = blockIdx.x * 16;
  const int K    = cols1 + cols2;
  const int half = lane >> 4;
  const int nlo  = lane & 15;

  // stage gathered/concatenated A tile into LDS as bf16
  for (int i = tid; i < 16 * K; i += 256) {
    int r = i / K, k = i - r * K;
    int m = m0 + r;
    float v = 0.0f;
    if (m < M) {
      if (k < cols1) {
        if (A1) {
          int rr = idx1 ? idx1[m] : m;
          v = A1[(size_t)rr * cols1 + k];
        }
      } else {
        int rr = idx2 ? idx2[m] : m;
        v = A2[(size_t)rr * cols2 + (k - cols1)];
      }
    }
    Ash[r * K + k] = (__bf16)v;
  }
  __syncthreads();

  const int ksteps = K >> 5;
  int  n0s[NT];
  bool val[NT];
  v8f  acc[NT];
#pragma unroll
  for (int j = 0; j < NT; ++j) {
    n0s[j] = (w + 8 * j) * 16;
    val[j] = n0s[j] < N;  // wave-uniform
    float bv = val[j] ? bias[n0s[j] + nlo] : 0.0f;
    acc[j] = {bv, bv, bv, bv, bv, bv, bv, bv};
  }
  for (int kk = 0; kk < ksteps; ++kk) {
    v16bf a = a_frag_lds_bf(Ash, K, kk * 32, lane);
    v16bf bfr[NT];
#pragma unroll
    for (int j = 0; j < NT; ++j)
      if (val[j]) bfr[j] = b_frag_w(W, K, n0s[j], kk * 32, lane);
#pragma unroll
    for (int j = 0; j < NT; ++j)
      if (val[j])
        acc[j] = __builtin_amdgcn_wmma_f32_16x16x32_bf16(false, a, false, bfr[j], (short)0,
                                                         acc[j], false, false);
  }

  if (mode == 0) {
#pragma unroll
    for (int j = 0; j < NT; ++j) {
      if (val[j]) {
#pragma unroll
        for (int i = 0; i < 8; ++i) {
          int m = m0 + i + 8 * half;
          if (m < M) Out[(size_t)m * N + n0s[j] + nlo] = acc[j][i];
        }
      }
    }
  } else {
#pragma unroll
    for (int j = 0; j < NT; ++j) {
#pragma unroll
      for (int i = 0; i < 8; ++i) outsh[(i + 8 * half) * HDIM + n0s[j] + nlo] = acc[j][i];
    }
    __syncthreads();
    if (tid < 16) {
      float s = 0.0f;
      for (int c = 0; c < HDIM; ++c) s += outsh[tid * HDIM + c];
      float mu = s * (1.0f / HDIM);
      float v = 0.0f;
      for (int c = 0; c < HDIM; ++c) {
        float d = outsh[tid * HDIM + c] - mu;
        v += d * d;
      }
      mu_s[tid] = mu;
      rs_s[tid] = rsqrtf(v * (1.0f / HDIM) + 1e-5f);
    }
    __syncthreads();
    const int c = tid;  // 256 threads <-> 256 channels, coalesced stores/atomics
    const float gc = gamma[c], bc = beta[c];
    for (int r = 0; r < 16; ++r) {
      int m = m0 + r;
      if (m >= M) continue;
      float v = (outsh[r * HDIM + c] - mu_s[r]) * rs_s[r] * gc + bc;
      v = fmaxf(v, 0.0f);
      if (mode == 1) {
        Out[(size_t)m * HDIM + c] = v;
      } else {
        int hi = histIdx[m];
        float msg = fmaxf(v - hist[(size_t)hi * HDIM + c], 0.0f);
        int d = dstIdx[m];
        atomicAdd(Out + (size_t)d * HDIM + c, msg);
      }
    }
  }
}

// ---------------------------------------------------------------------------
// Elementwise combiners
// ---------------------------------------------------------------------------
__global__ void newh0_kernel(const float* __restrict__ delta, const float* __restrict__ selfh,
                             const float* __restrict__ subg, float* __restrict__ out, int n) {
  int i = blockIdx.x * 256 + threadIdx.x;
  if (i >= n) return;
  int r = i >> 8;
  out[i] = (delta[i] - selfh[i]) * subg[r];
}

__global__ void newh1_kernel(const float* __restrict__ delta, const float* __restrict__ selfh,
                             const float* __restrict__ hist1, const int* __restrict__ self_nid,
                             const float* __restrict__ subg, const float* __restrict__ nrm,
                             const float* __restrict__ aggh, float* __restrict__ out, int n) {
  int i = blockIdx.x * 256 + threadIdx.x;
  if (i >= n) return;
  int r = i >> 8, c = i & 255;
  float sdelta = selfh[i] - hist1[(size_t)self_nid[r] * HDIM + c];
  out[i] = (delta[i] - sdelta) * subg[r] + aggh[i] * nrm[r];
}

// ---------------------------------------------------------------------------
// Host orchestration
// ---------------------------------------------------------------------------
extern "C" void kernel_launch(void* const* d_in, const int* in_sizes, int n_in,
                              void* d_out, int out_size, void* d_ws, size_t ws_size,
                              hipStream_t stream) {
  const float* node_features = (const float*)d_in[0];
  const float* ef0   = (const float*)d_in[1];
  const float* ef1   = (const float*)d_in[2];
  const float* hist0 = (const float*)d_in[3];
  const float* hist1 = (const float*)d_in[4];
  const float* aggh1 = (const float*)d_in[5];
  const float* subg1 = (const float*)d_in[6];
  const float* subg2 = (const float*)d_in[7];
  const float* nrm2  = (const float*)d_in[8];
  const int* src0 = (const int*)d_in[9];
  const int* dst0 = (const int*)d_in[10];
  const int* src1 = (const int*)d_in[11];
  const int* dst1 = (const int*)d_in[12];
  const int* self_nid0 = (const int*)d_in[13];
  const int* self_nid1 = (const int*)d_in[14];
  const float* g0Wih = (const float*)d_in[15];
  const float* g0Whh = (const float*)d_in[16];
  const float* g0bih = (const float*)d_in[17];
  const float* g0bhh = (const float*)d_in[18];
  const float* g1Wih = (const float*)d_in[19];
  const float* g1Whh = (const float*)d_in[20];
  const float* g1bih = (const float*)d_in[21];
  const float* g1bhh = (const float*)d_in[22];
  const float* phi0W = (const float*)d_in[23];
  const float* phi0b = (const float*)d_in[24];
  const float* phi0g = (const float*)d_in[25];
  const float* phi0be = (const float*)d_in[26];
  const float* phi1W = (const float*)d_in[27];
  const float* phi1b = (const float*)d_in[28];
  const float* phi1g = (const float*)d_in[29];
  const float* phi1be = (const float*)d_in[30];
  const float* nu0W = (const float*)d_in[31];
  const float* nu0b = (const float*)d_in[32];
  const float* nu0g = (const float*)d_in[33];
  const float* nu0be = (const float*)d_in[34];
  const float* nu1W = (const float*)d_in[35];
  const float* nu1b = (const float*)d_in[36];
  const float* nu1g = (const float*)d_in[37];
  const float* nu1be = (const float*)d_in[38];
  const float* fc1W = (const float*)d_in[39];
  const float* fc1b = (const float*)d_in[40];
  const float* fc2W = (const float*)d_in[41];
  const float* fc2b = (const float*)d_in[42];

  const int E0 = in_sizes[9];
  const int E1 = in_sizes[11];
  const int N1 = in_sizes[13];
  const int N2 = in_sizes[14];
  const int De = in_sizes[15] / (3 * HDIM);
  const int T  = in_sizes[1] / (E0 * De);
  const int N0 = in_sizes[3] / HDIM;
  const int Dn = in_sizes[0] / N0;
  const int FC = in_sizes[40];
  const int C  = in_sizes[42];

  // ---- carve workspace ----
  char* wsp = (char*)d_ws;
  size_t off = 0;
  auto alloc = [&](size_t bytes) -> void* {
    void* p = (void*)(wsp + off);
    off += (bytes + 255) & ~(size_t)255;
    return p;
  };
  __bf16* w_g0ih = (__bf16*)alloc((size_t)3 * HDIM * De * 2);
  __bf16* w_g0hh = (__bf16*)alloc((size_t)3 * HDIM * HDIM * 2);
  __bf16* w_g1ih = (__bf16*)alloc((size_t)3 * HDIM * HDIM * 2);
  __bf16* w_g1hh = (__bf16*)alloc((size_t)3 * HDIM * HDIM * 2);
  __bf16* w_phi0 = (__bf16*)alloc((size_t)HDIM * (Dn + HDIM) * 2);
  __bf16* w_phi1 = (__bf16*)alloc((size_t)HDIM * (2 * HDIM) * 2);
  __bf16* w_nu0  = (__bf16*)alloc((size_t)HDIM * (2 * HDIM) * 2);
  __bf16* w_nu1  = (__bf16*)alloc((size_t)HDIM * (2 * HDIM) * 2);
  __bf16* w_fc1  = (__bf16*)alloc((size_t)FC * HDIM * 2);
  __bf16* w_fc2  = (__bf16*)alloc((size_t)C * FC * 2);
  float* emb0   = (float*)alloc((size_t)E0 * HDIM * 4);
  float* selfh0 = (float*)alloc((size_t)N1 * HDIM * 4);
  float* delta0 = (float*)alloc((size_t)N1 * HDIM * 4);
  float* newh0  = (float*)alloc((size_t)N1 * HDIM * 4);
  float* h1     = (float*)alloc((size_t)N1 * HDIM * 4);
  float* emb1   = (float*)alloc((size_t)E1 * HDIM * 4);
  float* selfh1 = (float*)alloc((size_t)N2 * HDIM * 4);
  float* delta1 = (float*)alloc((size_t)N2 * HDIM * 4);
  float* newh1  = (float*)alloc((size_t)N2 * HDIM * 4);
  float* h2     = (float*)alloc((size_t)N2 * HDIM * 4);
  float* fc1o   = (float*)alloc((size_t)N2 * FC * 4);
  (void)ws_size; (void)n_in; (void)out_size;

  auto cvt = [&](const float* s, __bf16* dptr, int n) {
    cvt_bf16_kernel<<<(n + 255) / 256, 256, 0, stream>>>(s, dptr, n);
  };
  cvt(g0Wih, w_g0ih, 3 * HDIM * De);
  cvt(g0Whh, w_g0hh, 3 * HDIM * HDIM);
  cvt(g1Wih, w_g1ih, 3 * HDIM * HDIM);
  cvt(g1Whh, w_g1hh, 3 * HDIM * HDIM);
  cvt(phi0W, w_phi0, HDIM * (Dn + HDIM));
  cvt(phi1W, w_phi1, HDIM * 2 * HDIM);
  cvt(nu0W,  w_nu0,  HDIM * 2 * HDIM);
  cvt(nu1W,  w_nu1,  HDIM * 2 * HDIM);
  cvt(fc1W,  w_fc1,  FC * HDIM);
  cvt(fc2W,  w_fc2,  C * FC);

  auto gemm256 = [&](const float* A1, const int* i1, int c1,
                     const float* A2, const int* i2, int c2,
                     const __bf16* Wp, const float* bp, int M, int mode,
                     const float* gp, const float* bep,
                     const float* hp, const int* hxp, const int* dp, float* Op) {
    gemm_ln_kernel<2><<<(M + 15) / 16, 256, 0, stream>>>(A1, i1, c1, A2, i2, c2, Wp, bp, M,
                                                         HDIM, mode, gp, bep, hp, hxp, dp, Op);
  };

  // ---- layer 0 ----
  gru_mean_kernel<<<(E0 + 15) / 16, 512, 0, stream>>>(ef0, E0, T, De, w_g0ih, w_g0hh,
                                                      g0bih, g0bhh, emb0);
  // self_h0 = slnn(concat[zeros(H), node_features[self_nid0]])
  gemm256(nullptr, nullptr, HDIM, node_features, self_nid0, Dn, w_phi0, phi0b, N1, 1,
          phi0g, phi0be, nullptr, nullptr, nullptr, selfh0);
  zero_f32_kernel<<<(N1 * HDIM + 255) / 256, 256, 0, stream>>>(delta0, N1 * HDIM);
  // nb0 = slnn(concat[node_features[src0], emb0]); msg0 scatter-add into delta0
  gemm256(node_features, src0, Dn, emb0, nullptr, HDIM, w_phi0, phi0b, E0, 2,
          phi0g, phi0be, hist0, src0, dst0, delta0);
  newh0_kernel<<<(N1 * HDIM + 255) / 256, 256, 0, stream>>>(delta0, selfh0, subg1, newh0,
                                                            N1 * HDIM);
  gemm256(newh0, nullptr, HDIM, selfh0, nullptr, HDIM, w_nu0, nu0b, N1, 1,
          nu0g, nu0be, nullptr, nullptr, nullptr, h1);

  // ---- layer 1 ----
  gru_mean_kernel<<<(E1 + 15) / 16, 512, 0, stream>>>(ef1, E1, T, HDIM, w_g1ih, w_g1hh,
                                                      g1bih, g1bhh, emb1);
  gemm256(nullptr, nullptr, HDIM, h1, self_nid1, HDIM, w_phi1, phi1b, N2, 1,
          phi1g, phi1be, nullptr, nullptr, nullptr, selfh1);
  zero_f32_kernel<<<(N2 * HDIM + 255) / 256, 256, 0, stream>>>(delta1, N2 * HDIM);
  gemm256(h1, src1, HDIM, emb1, nullptr, HDIM, w_phi1, phi1b, E1, 2,
          phi1g, phi1be, hist1, src1, dst1, delta1);
  newh1_kernel<<<(N2 * HDIM + 255) / 256, 256, 0, stream>>>(delta1, selfh1, hist1, self_nid1,
                                                            subg2, nrm2, aggh1, newh1,
                                                            N2 * HDIM);
  gemm256(newh1, nullptr, HDIM, selfh1, nullptr, HDIM, w_nu1, nu1b, N2, 1,
          nu1g, nu1be, nullptr, nullptr, nullptr, h2);

  // ---- readout (no activation between fc1 and fc2) ----
  gemm_ln_kernel<4><<<(N2 + 15) / 16, 256, 0, stream>>>(
      h2, nullptr, HDIM, nullptr, nullptr, 0, w_fc1, fc1b, N2, FC, 0,
      nullptr, nullptr, nullptr, nullptr, nullptr, fc1o);
  gemm_ln_kernel<1><<<(N2 + 15) / 16, 256, 0, stream>>>(
      fc1o, nullptr, FC, nullptr, nullptr, 0, w_fc2, fc2b, N2, C, 0,
      nullptr, nullptr, nullptr, nullptr, nullptr, (float*)d_out);
}
